// LocalAttention_45406394254098
// MI455X (gfx1250) — matrix operand order repaired and checked by reading, not compiled, not run on hardware
//
#include <hip/hip_runtime.h>
#include <hip/hip_bf16.h>

typedef __attribute__((ext_vector_type(16))) __bf16 v16bf;
typedef __attribute__((ext_vector_type(4)))  __bf16 v4bf;
typedef __attribute__((ext_vector_type(8)))  float  v8f;

#define BATCH 2
#define SEQ   2048
#define MD    1024
#define NH    16
#define HD    64
#define WIN   256

// ---------------- fragment helpers (contiguous-k WMMA operand order used here) ----------------
// A 16x32 bf16: lane holds row M=lane&15; 16 contiguous K values starting at
// k0+16*(lane>>4). -> two 16B chunks at p and p+8.
__device__ inline v16bf ldfragA(const __bf16* p) {
    v16bf r;
    ((uint4*)&r)[0] = *(const uint4*)(p);
    ((uint4*)&r)[1] = *(const uint4*)(p + 8);
    return r;
}
// B 32x16 bf16: lane holds column N=lane&15; 16 contiguous K values starting at
// k0+16*(lane>>4). -> 32 contiguous bytes at p.
__device__ inline v16bf ldfragB(const __bf16* p) {
    v16bf r;
    ((uint4*)&r)[0] = *(const uint4*)(p);
    ((uint4*)&r)[1] = *(const uint4*)(p + 8);
    return r;
}

__device__ inline v8f wmma_bf16(v16bf a, v16bf b, v8f c) {
    return __builtin_amdgcn_wmma_f32_16x16x32_bf16(false, a, false, b, (short)0, c,
                                                   false, false);
}

// ---------------- staging kernels ----------------
// float4-vectorized f32 -> bf16 (8B stores)
__global__ void cvt_bf16_kernel(const float* __restrict__ in,
                                __bf16* __restrict__ out, int n4) {
    int i = blockIdx.x * blockDim.x + threadIdx.x;
    if (i < n4) {
        float4 f = ((const float4*)in)[i];
        v4bf o;
        o[0] = (__bf16)f.x; o[1] = (__bf16)f.y;
        o[2] = (__bf16)f.z; o[3] = (__bf16)f.w;
        *(v4bf*)(out + 4 * i) = o;
    }
}

// values (B,S,MD) f32 -> vT (B,NH,HD,SEQ) bf16, LDS-tiled 32x32 transpose:
// coalesced reads along d, coalesced writes along s.
__global__ void __launch_bounds__(256)
vtrans_kernel(const float* __restrict__ v, __bf16* __restrict__ vt) {
    __shared__ float tile[32][33];                  // +1 pad: no bank conflicts
    const int tx = threadIdx.x & 31;
    const int ty = threadIdx.x >> 5;                // 0..7
    const int s0 = blockIdx.x * 32;
    const int d0 = blockIdx.y * 32;
    const int bh = blockIdx.z;                      // b*NH + h
    const int b  = bh >> 4, h = bh & 15;

    const float* src = v + (size_t)(b * SEQ + s0) * MD + h * HD + d0;
#pragma unroll
    for (int i = 0; i < 4; ++i)
        tile[ty + i * 8][tx] = src[(size_t)(ty + i * 8) * MD + tx];
    __syncthreads();
    __bf16* dst = vt + (size_t)(bh * HD + d0) * SEQ + s0;
#pragma unroll
    for (int i = 0; i < 4; ++i)
        dst[(size_t)(ty + i * 8) * SEQ + tx] = (__bf16)tile[tx][ty + i * 8];
}

// ---------------- generic  C = A (MxK) * W^T  (W is NxK row-major) ------------
template <typename OutT>
__global__ void __launch_bounds__(256)
gemm_xwT_kernel(const __bf16* __restrict__ A, const __bf16* __restrict__ W,
                OutT* __restrict__ C, int Mrows, int K, int N) {
    const int lane = threadIdx.x & 31;
    const int w    = threadIdx.x >> 5;
    const int col  = lane & 15;
    const int half = lane >> 4;
    const int m0   = blockIdx.y * 128 + w * 16;
    const int n0   = blockIdx.x * 64;

    v8f acc[4] = {v8f{}, v8f{}, v8f{}, v8f{}};
    const int arow = m0 + col;

    for (int k0 = 0; k0 < K; k0 += 32) {
        v16bf a = ldfragA(A + (size_t)arow * K + k0 + 16 * half);
#pragma unroll
        for (int t = 0; t < 4; ++t) {
            v16bf bw = ldfragB(W + (size_t)(n0 + t * 16 + col) * K + k0 + 16 * half);
            acc[t] = wmma_bf16(a, bw, acc[t]);
        }
    }
#pragma unroll
    for (int r = 0; r < 8; ++r) {
        int m = m0 + r + 8 * half;
#pragma unroll
        for (int t = 0; t < 4; ++t) {
            C[(size_t)m * N + n0 + t * 16 + col] = (OutT)acc[t][r];
        }
    }
}

// ---------------- fused sliding-window attention ------------------------------
// One wave per (b, h, 16-query tile). Flash-style online softmax.
// Row-sum l is accumulated as a 5th WMMA column (P x ones) instead of shuffles.
__global__ void __launch_bounds__(128)
attn_kernel(const __bf16* __restrict__ Q,   // (B*S, MD) head-major columns
            const __bf16* __restrict__ Kb,  // (B*S, MD)
            const __bf16* __restrict__ Vt,  // (B*NH*HD, SEQ)
            __bf16* __restrict__ O) {       // (B*S, MD)
    __shared__ __bf16 plds[4][16 * 32];     // P bounce buffer, 1KB per wave

    const int lane = threadIdx.x & 31;
    const int w    = threadIdx.x >> 5;
    const int wid  = blockIdx.x * 4 + w;    // b*NH*(SEQ/16) + h*(SEQ/16) + qt
    const int qt   = wid & 127;             // SEQ/16 = 128
    const int h    = (wid >> 7) & (NH - 1);
    const int b    = wid >> 11;
    const int qs0  = qt * 16;
    const int col  = lane & 15;
    const int half = lane >> 4;

    // Q tile as two A-fragments (d=0..31, d=32..63)
    const int qrow = qs0 + col;
    const __bf16* qp = Q + (size_t)(b * SEQ + qrow) * MD + h * HD;
    v16bf aq0 = ldfragA(qp + 0  + 16 * half);
    v16bf aq1 = ldfragA(qp + 32 + 16 * half);

    // all-ones B fragment: accumulates row sums of P via WMMA
    v16bf ones;
#pragma unroll
    for (int i = 0; i < 16; ++i) ones[i] = (__bf16)1.0f;

    v8f o0 = v8f{}, o1 = v8f{}, o2 = v8f{}, o3 = v8f{};  // 16x64 f32 accumulator
    v8f lsum = v8f{};                                    // running row-sum of P
    float m_run[8];
#pragma unroll
    for (int r = 0; r < 8; ++r) m_run[r] = -3.0e38f;

    const int kb_lo = (qs0 > WIN) ? ((qs0 - WIN) & ~31) : 0;
    for (int kb = kb_lo; kb < qs0 + 16; kb += 32) {
        // prefetch next key block's K rows and V lines (global_prefetch_b8)
        if (kb + 32 < qs0 + 16) {
            int pr = kb + 32 + lane;
            if (pr > SEQ - 1) pr = SEQ - 1;
            __builtin_prefetch(Kb + (size_t)(b * SEQ + pr) * MD + h * HD, 0, 0);
            __builtin_prefetch(
                Vt + (size_t)((b * NH + h) * HD + lane * 2) * SEQ + kb + 32, 0, 0);
        }

        // ---- scores: two 16x16 tiles over 32 keys, contraction d=64 ----
        v8f sc[2];
#pragma unroll
        for (int sub = 0; sub < 2; ++sub) {
            const int krow = kb + sub * 16 + col;
            const __bf16* kp = Kb + (size_t)(b * SEQ + krow) * MD + h * HD;
            v16bf bk0 = ldfragB(kp + 0  + 16 * half);
            v16bf bk1 = ldfragB(kp + 32 + 16 * half);
            v8f c = v8f{};
            c = wmma_bf16(aq0, bk0, c);
            c = wmma_bf16(aq1, bk1, c);
            sc[sub] = c;
        }
        // ---- mask + scale + online max + P scatter ----
        const int kp0 = kb + col, kp1 = kb + 16 + col;
#pragma unroll
        for (int r = 0; r < 8; ++r) {
            const int qpos = qs0 + r + 8 * half;
            float s0 = sc[0][r] * 0.125f;           // 1/sqrt(64)
            float s1 = sc[1][r] * 0.125f;
            const int d0 = qpos - kp0, d1 = qpos - kp1;
            s0 = (d0 >= 1 && d0 <= WIN) ? s0 : -1e9f;
            s1 = (d1 >= 1 && d1 <= WIN) ? s1 : -1e9f;

            float mx = fmaxf(s0, s1);               // row-max across 16 lanes
#pragma unroll
            for (int off = 1; off < 16; off <<= 1)
                mx = fmaxf(mx, __shfl_xor(mx, off, 32));

            const float mnew = fmaxf(m_run[r], mx);
            const float corr = __expf(m_run[r] - mnew);
            m_run[r] = mnew;
            o0[r] *= corr; o1[r] *= corr; o2[r] *= corr; o3[r] *= corr;
            lsum[r] *= corr;

            const float p0 = __expf(s0 - mnew);
            const float p1 = __expf(s1 - mnew);

            // scatter P into LDS as a row-major 16x32 bf16 tile
            const int row = r + 8 * half;
            plds[w][row * 32 + col]      = (__bf16)p0;
            plds[w][row * 32 + 16 + col] = (__bf16)p1;
        }
        // cross-lane LDS RAW: wait for DS stores, block compiler reordering
        asm volatile("s_wait_dscnt 0" ::: "memory");

        // re-load P as an A-fragment (16x32); O += P*V, lsum += P*1
        v16bf ap = ldfragA(&plds[w][col * 32 + 16 * half]);
        const __bf16* vbase = Vt + (size_t)((b * NH + h) * HD + col) * SEQ + kb + 16 * half;
        o0 = wmma_bf16(ap, ldfragB(vbase + (size_t)(0 * 16) * SEQ), o0);
        o1 = wmma_bf16(ap, ldfragB(vbase + (size_t)(1 * 16) * SEQ), o1);
        o2 = wmma_bf16(ap, ldfragB(vbase + (size_t)(2 * 16) * SEQ), o2);
        o3 = wmma_bf16(ap, ldfragB(vbase + (size_t)(3 * 16) * SEQ), o3);
        lsum = wmma_bf16(ap, ones, lsum);
    }

    // ---- normalize + store (row 0 has no valid keys -> zero) ----
#pragma unroll
    for (int r = 0; r < 8; ++r) {
        const int qpos = qs0 + r + 8 * half;
        const float l   = lsum[r];
        const float inv = (qpos >= 1 && l > 0.f) ? (1.0f / l) : 0.0f;
        __bf16* op = O + (size_t)(b * SEQ + qpos) * MD + h * HD + col;
        op[0]  = (__bf16)(o0[r] * inv);
        op[16] = (__bf16)(o1[r] * inv);
        op[32] = (__bf16)(o2[r] * inv);
        op[48] = (__bf16)(o3[r] * inv);
    }
}

// ---------------- host launcher ----------------
extern "C" void kernel_launch(void* const* d_in, const int* in_sizes, int n_in,
                              void* d_out, int out_size, void* d_ws, size_t ws_size,
                              hipStream_t stream) {
    const float* query = (const float*)d_in[0];
    const float* keys  = (const float*)d_in[1];
    const float* vals  = (const float*)d_in[2];
    const float* Wq    = (const float*)d_in[3];
    const float* Wo    = (const float*)d_in[4];
    // window=256, topk=0 hardcoded per reference setup

    const int NACT = BATCH * SEQ * MD;   // 4M elements
    const int NW   = MD * MD;            // 1M elements

    __bf16* qin = (__bf16*)d_ws;         // staged query        4M
    __bf16* kin = qin + NACT;            // staged keys         4M
    __bf16* vT  = kin + NACT;            // transposed values   4M
    __bf16* wq  = vT  + NACT;            // staged Wq           1M
    __bf16* wo  = wq  + NW;              // staged Wo           1M
    __bf16* qpr = wo  + NW;              // projected Q         4M
    __bf16* att = qpr + NACT;            // attention output    4M   (44 MB total)

    cvt_bf16_kernel<<<(NACT / 4) / 256, 256, 0, stream>>>(query, qin, NACT / 4);
    cvt_bf16_kernel<<<(NACT / 4) / 256, 256, 0, stream>>>(keys, kin, NACT / 4);
    vtrans_kernel<<<dim3(SEQ / 32, HD / 32, BATCH * NH), 256, 0, stream>>>(vals, vT);
    cvt_bf16_kernel<<<(NW / 4) / 256, 256, 0, stream>>>(Wq, wq, NW / 4);
    cvt_bf16_kernel<<<(NW / 4) / 256, 256, 0, stream>>>(Wo, wo, NW / 4);

    // Q projection: (4096x1024) = query_bf * Wq^T   -> bf16
    gemm_xwT_kernel<__bf16><<<dim3(MD / 64, (BATCH * SEQ) / 128), 256, 0, stream>>>(
        qin, wq, qpr, BATCH * SEQ, MD, MD);

    // fused sliding-window attention: 4096 waves, 4 per block
    attn_kernel<<<(BATCH * NH * (SEQ / 16)) / 4, 128, 0, stream>>>(qpr, kin, vT, att);

    // output projection: (4096x1024) = attn_bf * Wo^T -> f32
    gemm_xwT_kernel<float><<<dim3(MD / 64, (BATCH * SEQ) / 128), 256, 0, stream>>>(
        att, wo, (float*)d_out, BATCH * SEQ, MD, MD);
}
